// GAT_67740224193270
// MI455X (gfx1250) — compile-verified, matrix operand order checked by reading, not run on hardware
//
#include <hip/hip_runtime.h>
#include <hip/hip_bf16.h>

typedef __attribute__((ext_vector_type(16))) _Float16 v16h;
typedef __attribute__((ext_vector_type(8)))  _Float16 v8h;
typedef __attribute__((ext_vector_type(8)))  float    v8f;

#define NEG_SLOPE 0.2f

// ---------------------------------------------------------------------------
// f32 -> f16 elementwise convert
// ---------------------------------------------------------------------------
__global__ void cvt_f16_kernel(const float* __restrict__ in,
                               _Float16* __restrict__ out, int n) {
  int i = blockIdx.x * blockDim.x + threadIdx.x;
  if (i < n) out[i] = (_Float16)in[i];
}

// ---------------------------------------------------------------------------
// Pack W [K, Nout] (row-major f32) into per-lane WMMA B-fragment order (f16).
// tid = (nt*KS + ks)*32 + lane ; 16 halves per tid, contiguous.
// Lane l (l<16): column n = nt*16+l, halves j=0..7 -> K = ks*32 + (l>>4)*8 + j,
// halves 8..15 -> +16.  Mirrors the 16-bit A-matrix layout (ISA 7.12.2).
// ---------------------------------------------------------------------------
__global__ void pack_w_kernel(const float* __restrict__ W,
                              _Float16* __restrict__ Wp, int K, int Nout) {
  int tid = blockIdx.x * blockDim.x + threadIdx.x;
  int KS = K >> 5;
  int NT = Nout >> 4;
  if (tid >= NT * KS * 32) return;
  int lane = tid & 31;
  int tmp  = tid >> 5;
  int ks   = tmp % KS;
  int nt   = tmp / KS;
  int r = lane & 15, half = lane >> 4;
  int n  = nt * 16 + r;
  int kb = ks * 32;
  _Float16* dst = Wp + (size_t)tid * 16;
#pragma unroll
  for (int j = 0; j < 8; ++j)
    dst[j] = (_Float16)W[(size_t)(kb + half * 8 + j) * Nout + n];
#pragma unroll
  for (int j = 0; j < 8; ++j)
    dst[8 + j] = (_Float16)W[(size_t)(kb + 16 + half * 8 + j) * Nout + n];
}

// ---------------------------------------------------------------------------
// CSR row pointers from sorted dst: row_ptr[i] = lower_bound(dst, i)
// ---------------------------------------------------------------------------
__global__ void row_ptr_kernel(const int* __restrict__ dst,
                               int* __restrict__ row_ptr, int N, int E) {
  int i = blockIdx.x * blockDim.x + threadIdx.x;
  if (i > N) return;
  int lo = 0, hi = E;
  while (lo < hi) {
    int mid = (lo + hi) >> 1;
    if (dst[mid] < i) lo = mid + 1; else hi = mid;
  }
  row_ptr[i] = lo;
}

// ---------------------------------------------------------------------------
// GEMM C[M,Nout] = A[M,K] * B[K,Nout] using v_wmma_f32_16x16x32_f16.
// 128 threads = 4 waves; each wave owns TPW 16-wide column tiles of a
// 16-row slab. A loaded straight from row-major f16 (two 16B chunks/lane),
// B from the pre-packed fragment buffer.
// ---------------------------------------------------------------------------
template <int TPW>
__global__ void gemm_wmma_f16(const _Float16* __restrict__ A,
                              const _Float16* __restrict__ Bp,
                              float* __restrict__ C, int M, int K, int Nout) {
  int wave = threadIdx.x >> 5;
  int lane = threadIdx.x & 31;
  int r = lane & 15, half = lane >> 4;
  int m0 = blockIdx.x * 16;
  int KS = K >> 5;
  int rowA = m0 + r;
  if (rowA >= M) rowA = M - 1;

  v8f acc[TPW];
#pragma unroll
  for (int t = 0; t < TPW; ++t)
#pragma unroll
    for (int i = 0; i < 8; ++i) acc[t][i] = 0.f;

  for (int ks = 0; ks < KS; ++ks) {
    const _Float16* ap = A + (size_t)rowA * K + ks * 32 + half * 8;
    v8h alo = *(const v8h*)ap;
    v8h ahi = *(const v8h*)(ap + 16);
    v16h a;
#pragma unroll
    for (int j = 0; j < 8; ++j) { a[j] = alo[j]; a[8 + j] = ahi[j]; }
#pragma unroll
    for (int t = 0; t < TPW; ++t) {
      int nt = wave * TPW + t;
      const _Float16* bp = Bp + ((size_t)(nt * KS + ks) * 32 + lane) * 16;
      v8h blo = *(const v8h*)bp;
      v8h bhi = *(const v8h*)(bp + 8);
      v16h b;
#pragma unroll
      for (int j = 0; j < 8; ++j) { b[j] = blo[j]; b[8 + j] = bhi[j]; }
      acc[t] = __builtin_amdgcn_wmma_f32_16x16x32_f16(
          false, a, false, b, (short)0, acc[t], false, false);
    }
  }

#pragma unroll
  for (int t = 0; t < TPW; ++t) {
    int n0 = (wave * TPW + t) * 16;
#pragma unroll
    for (int i = 0; i < 8; ++i) {
      int row = m0 + half * 8 + i;  // C layout: lanes 0-15 -> M=i, 16-31 -> M=8+i
      if (row < M) C[(size_t)row * Nout + n0 + r] = acc[t][i];
    }
  }
}

// ---------------------------------------------------------------------------
// el[n,h] = <feat[n,h,:], al[h,:]> ; er likewise (one feat read for both)
// ---------------------------------------------------------------------------
__global__ void attn_coeff_kernel(const float* __restrict__ feat,
                                  const float* __restrict__ al,
                                  const float* __restrict__ ar,
                                  float* __restrict__ el, float* __restrict__ er,
                                  int N, int H, int D) {
  int tid = blockIdx.x * blockDim.x + threadIdx.x;
  if (tid >= N * H) return;
  int h = tid % H;
  int n = tid / H;
  const float* f   = feat + (size_t)n * H * D + h * D;
  const float* alp = al + h * D;
  const float* arp = ar + h * D;
  float sl = 0.f, sr = 0.f;
  for (int d = 0; d < D; ++d) { float v = f[d]; sl += v * alp[d]; sr += v * arp[d]; }
  el[tid] = sl;
  er[tid] = sr;
}

// ---------------------------------------------------------------------------
// Per-dst-node edge softmax + aggregation. One block per node, thread t owns
// output channel t (head h = t/D). Pass 1: online softmax (max+sum) over
// incoming edges; pass 2: weighted sum of feat[src]. Deterministic (no atomics,
// dst-sorted CSR). FINAL: write f32 logits; else ELU -> f16 next-layer input.
// ---------------------------------------------------------------------------
template <bool FINAL>
__global__ void gat_aggregate(const float* __restrict__ feat,
                              const float* __restrict__ el,
                              const float* __restrict__ er,
                              const float* __restrict__ bias,
                              const int* __restrict__ src,
                              const int* __restrict__ row_ptr,
                              int H, int D,
                              _Float16* __restrict__ out16,
                              float* __restrict__ out32) {
  int n = blockIdx.x;
  int t = threadIdx.x;
  int DT = H * D;
  int h = t / D;
  float er_n = er[(size_t)n * H + h];
  int rs = row_ptr[n], re = row_ptr[n + 1];

  float m = -INFINITY, s = 0.f;
  for (int e = rs; e < re; ++e) {
    int si = src[e];
    float ev = el[(size_t)si * H + h] + er_n;
    ev = ev > 0.f ? ev : NEG_SLOPE * ev;
    if (ev > m) { s = s * __expf(m - ev) + 1.f; m = ev; }
    else         s += __expf(ev - m);
  }

  float acc = 0.f;
  for (int e = rs; e < re; ++e) {
    int si = src[e];
    float ev = el[(size_t)si * H + h] + er_n;
    ev = ev > 0.f ? ev : NEG_SLOPE * ev;
    acc += __expf(ev - m) * feat[(size_t)si * DT + t];
  }

  float res = acc / (s + 1e-9f) + bias[t];
  if (FINAL) {
    out32[(size_t)n * DT + t] = res;
  } else {
    res = res > 0.f ? res : (__expf(res) - 1.f);  // ELU
    out16[(size_t)n * DT + t] = (_Float16)res;
  }
}

// ---------------------------------------------------------------------------
extern "C" void kernel_launch(void* const* d_in, const int* in_sizes, int n_in,
                              void* d_out, int out_size, void* d_ws, size_t ws_size,
                              hipStream_t stream) {
  const float* x    = (const float*)d_in[0];
  const int*   src  = (const int*)d_in[1];
  const int*   dstI = (const int*)d_in[2];
  const float* W1   = (const float*)d_in[3];
  const float* al1  = (const float*)d_in[4];
  const float* ar1  = (const float*)d_in[5];
  const float* b1   = (const float*)d_in[6];
  const float* W2   = (const float*)d_in[7];
  const float* al2  = (const float*)d_in[8];
  const float* ar2  = (const float*)d_in[9];
  const float* b2   = (const float*)d_in[10];
  const float* W3   = (const float*)d_in[11];
  const float* al3  = (const float*)d_in[12];
  const float* ar3  = (const float*)d_in[13];
  const float* b3   = (const float*)d_in[14];
  float* out = (float*)d_out;

  const int N = in_sizes[0] / 256;
  const int E = in_sizes[1];

  // carve workspace (256B-aligned chunks)
  char* base = (char*)d_ws;
  size_t off = 0;
  auto carve = [&](size_t bytes) -> void* {
    void* p = base + off;
    off += (bytes + 255) & ~(size_t)255;
    return p;
  };
  _Float16* xh   = (_Float16*)carve((size_t)N * 256 * sizeof(_Float16));
  _Float16* h16  = (_Float16*)carve((size_t)N * 128 * sizeof(_Float16));
  float*    feat = (float*)   carve((size_t)N * 128 * sizeof(float));
  float*    el   = (float*)   carve((size_t)N * 4 * sizeof(float));
  float*    er   = (float*)   carve((size_t)N * 4 * sizeof(float));
  int*      rp   = (int*)     carve((size_t)(N + 1) * sizeof(int));
  _Float16* w1p  = (_Float16*)carve((size_t)8 * 8 * 32 * 16 * sizeof(_Float16)); // NT=8 KS=8
  _Float16* w2p  = (_Float16*)carve((size_t)8 * 4 * 32 * 16 * sizeof(_Float16)); // NT=8 KS=4
  _Float16* w3p  = (_Float16*)carve((size_t)4 * 4 * 32 * 16 * sizeof(_Float16)); // NT=4 KS=4

  const int thr = 256;
  // prep
  cvt_f16_kernel<<<(N * 256 + thr - 1) / thr, thr, 0, stream>>>(x, xh, N * 256);
  pack_w_kernel<<<(8 * 8 * 32 + thr - 1) / thr, thr, 0, stream>>>(W1, w1p, 256, 128);
  pack_w_kernel<<<(8 * 4 * 32 + thr - 1) / thr, thr, 0, stream>>>(W2, w2p, 128, 128);
  pack_w_kernel<<<(4 * 4 * 32 + thr - 1) / thr, thr, 0, stream>>>(W3, w3p, 128, 64);
  row_ptr_kernel<<<(N + 1 + thr - 1) / thr, thr, 0, stream>>>(dstI, rp, N, E);

  const int mblocks = (N + 15) / 16;

  // layer 1: 256 -> 4x32, ELU
  gemm_wmma_f16<2><<<mblocks, 128, 0, stream>>>(xh, w1p, feat, N, 256, 128);
  attn_coeff_kernel<<<(N * 4 + thr - 1) / thr, thr, 0, stream>>>(feat, al1, ar1, el, er, N, 4, 32);
  gat_aggregate<false><<<N, 128, 0, stream>>>(feat, el, er, b1, src, rp, 4, 32, h16, nullptr);

  // layer 2: 128 -> 4x32, ELU
  gemm_wmma_f16<2><<<mblocks, 128, 0, stream>>>(h16, w2p, feat, N, 128, 128);
  attn_coeff_kernel<<<(N * 4 + thr - 1) / thr, thr, 0, stream>>>(feat, al2, ar2, el, er, N, 4, 32);
  gat_aggregate<false><<<N, 128, 0, stream>>>(feat, el, er, b2, src, rp, 4, 32, h16, nullptr);

  // layer 3: 128 -> 1x64, no activation, mean over 1 head = identity
  gemm_wmma_f16<1><<<mblocks, 128, 0, stream>>>(h16, w3p, feat, N, 128, 64);
  attn_coeff_kernel<<<(N + thr - 1) / thr, thr, 0, stream>>>(feat, al3, ar3, el, er, N, 1, 64);
  gat_aggregate<true><<<N, 64, 0, stream>>>(feat, el, er, b3, src, rp, 1, 64, nullptr, out);
}
